// AttentionTransformer_51187420234304
// MI455X (gfx1250) — compile-verified, matrix operand order checked by reading, not run on hardware
//
#include <hip/hip_runtime.h>
#include <hip/hip_bf16.h>

#define B_ROWS   16384
#define D_A      512
#define INP_DIM  2048
#define VBS      1024
#define N_CHUNKS (B_ROWS / VBS)   // 16
#define EPS      1e-5f

typedef __bf16 bf16_t;
typedef bf16_t v16bf __attribute__((ext_vector_type(16)));
typedef float  v8f   __attribute__((ext_vector_type(8)));

// ---------------------------------------------------------------------------
// Async global -> LDS copy (CDNA5): 16 bytes per lane, tracked by ASYNCcnt.
// ---------------------------------------------------------------------------
__device__ __forceinline__ void async_copy_b128(unsigned lds_off,
                                                const void* gptr) {
    asm volatile("global_load_async_to_lds_b128 %0, %1, off"
                 :: "v"(lds_off), "v"(gptr)
                 : "memory");
}

// ---------------------------------------------------------------------------
// Kernel 1: f32 -> bf16 conversion (grid-stride)
// ---------------------------------------------------------------------------
__global__ void cvt_bf16_kernel(const float* __restrict__ src,
                                bf16_t* __restrict__ dst, int n) {
    for (int i = blockIdx.x * blockDim.x + threadIdx.x; i < n;
         i += gridDim.x * blockDim.x) {
        dst[i] = (bf16_t)src[i];
    }
}

// ---------------------------------------------------------------------------
// Kernel 2: H = A(bf16) @ W(bf16)^T + bias, f32 accumulate via WMMA.
// A: [16384 x 512] row-major (K contiguous)
// W: [2048  x 512] row-major (K contiguous) -> B-matrix columns
// Block: 256 threads = 8 waves as 2(M) x 4(N); wave tile 32x64.
// Workgroup tile: 64(M) x 256(N). Grid: (2048/256, 16384/64) = (8, 256).
// Tiles staged via double-buffered async global->LDS DMA; fragments read
// back with ds_load. LDS rows padded to 96 B (48 bf16) for bank spread.
// ---------------------------------------------------------------------------
#define LDS_ROW 48  // bf16 elements per LDS row (32 data + 16 pad)

__global__ __launch_bounds__(256) void gemm_bf16_wmma_kernel(
    const bf16_t* __restrict__ A, const bf16_t* __restrict__ Wm,
    const float* __restrict__ bias, float* __restrict__ H) {
    __shared__ __align__(64) bf16_t shA[2][64][LDS_ROW];    // 2 x 6 KB
    __shared__ __align__(64) bf16_t shB[2][256][LDS_ROW];   // 2 x 24 KB

    const int t     = threadIdx.x;
    const int lane  = t & 31;
    const int wave  = t >> 5;
    const int waveM = wave >> 2;   // 0..1
    const int waveN = wave & 3;    // 0..3
    const int m0g = blockIdx.y * 64;
    const int n0g = blockIdx.x * 256;
    const int lmod  = lane & 15;   // row (A) / col (B) within 16-tile
    const int lhalf = lane >> 4;   // K half selector

    v8f acc[2][4];
#pragma unroll
    for (int mi = 0; mi < 2; ++mi)
#pragma unroll
        for (int ni = 0; ni < 4; ++ni)
            acc[mi][ni] = (v8f){0.f, 0.f, 0.f, 0.f, 0.f, 0.f, 0.f, 0.f};

    // Issue the 5 async-DMA instructions (per wave) staging one k-slice:
    // A tile 64x32 (256 x 16B chunks), B tile 256x32 (1024 x 16B chunks).
    auto issue = [&](int buf, int k0) {
        {
            const int row = t >> 2, c = t & 3;
            async_copy_b128((unsigned)(size_t)&shA[buf][row][c * 8],
                            A + (size_t)(m0g + row) * D_A + k0 + c * 8);
        }
#pragma unroll
        for (int i = 0; i < 4; ++i) {
            const int cid = i * 256 + t;
            const int row = cid >> 2, c = cid & 3;
            async_copy_b128((unsigned)(size_t)&shB[buf][row][c * 8],
                            Wm + (size_t)(n0g + row) * D_A + k0 + c * 8);
        }
    };

    issue(0, 0);
    for (int it = 0; it < D_A / 32; ++it) {
        const int buf = it & 1;
        if (it + 1 < D_A / 32) {
            issue(buf ^ 1, (it + 1) * 32);
            // In-order completion: <=5 outstanding => current buffer landed.
            asm volatile("s_wait_asynccnt 0x5" ::: "memory");
        } else {
            asm volatile("s_wait_asynccnt 0x0" ::: "memory");
        }
        __syncthreads();  // all waves' DMA chunks visible

        v16bf af[2], bfr[4];
#pragma unroll
        for (int mi = 0; mi < 2; ++mi)
            af[mi] = *(const v16bf*)&shA[buf][waveM * 32 + mi * 16 + lmod]
                                        [lhalf * 16];
#pragma unroll
        for (int ni = 0; ni < 4; ++ni)
            bfr[ni] = *(const v16bf*)&shB[buf][waveN * 64 + ni * 16 + lmod]
                                         [lhalf * 16];
#pragma unroll
        for (int mi = 0; mi < 2; ++mi)
#pragma unroll
            for (int ni = 0; ni < 4; ++ni)
                acc[mi][ni] = __builtin_amdgcn_wmma_f32_16x16x32_bf16(
                    false, af[mi], false, bfr[ni], (short)0, acc[mi][ni],
                    false, false);
        __syncthreads();  // buffer safe to overwrite next iteration
    }

    // Epilogue: C/D layout — lanes 0-15: VGPR r <-> M=r; lanes 16-31: M=8+r.
    const int rowBase = m0g + waveM * 32 + (lhalf << 3);
#pragma unroll
    for (int mi = 0; mi < 2; ++mi) {
#pragma unroll
        for (int ni = 0; ni < 4; ++ni) {
            const int col = n0g + waveN * 64 + ni * 16 + lmod;
            const float bv = bias[col];
#pragma unroll
            for (int r = 0; r < 8; ++r) {
                H[(size_t)(rowBase + mi * 16 + r) * INP_DIM + col] =
                    acc[mi][ni][r] + bv;
            }
        }
    }
}

// ---------------------------------------------------------------------------
// Kernel 3: ghost-BN statistics. One thread per (chunk, col):
// mean/var over 1024 rows (biased variance, E[x^2]-m^2).
// ---------------------------------------------------------------------------
__global__ __launch_bounds__(256) void gbn_stats_kernel(
    const float* __restrict__ H, float* __restrict__ meanv,
    float* __restrict__ varv) {
    const int gid = blockIdx.x * blockDim.x + threadIdx.x;  // 0..32767
    const int chunk = gid >> 11;
    const int col = gid & (INP_DIM - 1);
    const float* p = H + (size_t)chunk * VBS * INP_DIM + col;
    float s = 0.f, ss = 0.f;
    for (int r = 0; r < VBS; ++r) {
        float v = p[(size_t)r * INP_DIM];
        s += v;
        ss += v * v;
    }
    const float m = s * (1.f / (float)VBS);
    meanv[gid] = m;
    varv[gid] = ss * (1.f / (float)VBS) - m * m;
}

// ---------------------------------------------------------------------------
// Kernel 4: per-row normalize * priors, then sparsemax (Michelot fixed
// point) in-place. One 256-thread block per row; 8 elements/thread.
// ---------------------------------------------------------------------------
__device__ __forceinline__ float wave_sum32(float v) {
#pragma unroll
    for (int off = 16; off > 0; off >>= 1) v += __shfl_xor(v, off, 32);
    return v;
}
__device__ __forceinline__ float wave_max32(float v) {
#pragma unroll
    for (int off = 16; off > 0; off >>= 1) v = fmaxf(v, __shfl_xor(v, off, 32));
    return v;
}

__global__ __launch_bounds__(256) void sparsemax_kernel(
    float* __restrict__ HO, const float* __restrict__ priors,
    const float* __restrict__ meanv, const float* __restrict__ varv,
    const float* __restrict__ gamma, const float* __restrict__ beta) {
    __shared__ float redA[8];
    __shared__ float redB[8];
    const int row = blockIdx.x;
    const int chunk = row >> 10;  // row / VBS
    const int t = threadIdx.x;
    const int lane = t & 31;
    const int wave = t >> 5;

    float z[8];
#pragma unroll
    for (int k = 0; k < 8; ++k) {
        const int j = t + k * 256;
        const float m = meanv[chunk * INP_DIM + j];
        const float v = varv[chunk * INP_DIM + j];
        const float hv = HO[(size_t)row * INP_DIM + j];
        const float y = (hv - m) * rsqrtf(v + EPS);
        z[k] = (y * gamma[j] + beta[j]) * priors[(size_t)row * INP_DIM + j];
    }

    // 1) subtract row max
    float mx = z[0];
#pragma unroll
    for (int k = 1; k < 8; ++k) mx = fmaxf(mx, z[k]);
    mx = wave_max32(mx);
    if (lane == 0) redA[wave] = mx;
    __syncthreads();
    mx = redA[0];
#pragma unroll
    for (int w = 1; w < 8; ++w) mx = fmaxf(mx, redA[w]);
    __syncthreads();
#pragma unroll
    for (int k = 0; k < 8; ++k) z[k] -= mx;

    // 2) initial tau from full support
    float s = 0.f;
#pragma unroll
    for (int k = 0; k < 8; ++k) s += z[k];
    s = wave_sum32(s);
    if (lane == 0) redA[wave] = s;
    __syncthreads();
    float total = 0.f;
#pragma unroll
    for (int w = 0; w < 8; ++w) total += redA[w];
    __syncthreads();
    float tau = (total - 1.f) * (1.f / (float)INP_DIM);

    // 3) Michelot iterations: tau = (sum_{z>tau} z - 1) / count
    for (int it = 0; it < 24; ++it) {
        float ps = 0.f, pc = 0.f;
#pragma unroll
        for (int k = 0; k < 8; ++k) {
            if (z[k] > tau) { ps += z[k]; pc += 1.f; }
        }
        ps = wave_sum32(ps);
        pc = wave_sum32(pc);
        if (lane == 0) { redA[wave] = ps; redB[wave] = pc; }
        __syncthreads();
        float sT = 0.f, cT = 0.f;
#pragma unroll
        for (int w = 0; w < 8; ++w) { sT += redA[w]; cT += redB[w]; }
        __syncthreads();
        if (cT > 0.f) tau = (sT - 1.f) / cT;
    }

    // 4) project and write
#pragma unroll
    for (int k = 0; k < 8; ++k) {
        const int j = t + k * 256;
        HO[(size_t)row * INP_DIM + j] = fmaxf(z[k] - tau, 0.f);
    }
}

// ---------------------------------------------------------------------------
extern "C" void kernel_launch(void* const* d_in, const int* in_sizes, int n_in,
                              void* d_out, int out_size, void* d_ws,
                              size_t ws_size, hipStream_t stream) {
    const float* a      = (const float*)d_in[0];   // [16384, 512]
    const float* priors = (const float*)d_in[1];   // [16384, 2048]
    const float* W      = (const float*)d_in[2];   // [2048, 512]
    const float* bias   = (const float*)d_in[3];   // [2048]
    const float* gamma  = (const float*)d_in[4];   // [2048]
    const float* beta   = (const float*)d_in[5];   // [2048]
    float* out = (float*)d_out;                    // [16384, 2048], used as H

    char* ws = (char*)d_ws;
    const size_t aBytes = (size_t)B_ROWS * D_A * sizeof(bf16_t);   // 16 MB
    const size_t wBytes = (size_t)INP_DIM * D_A * sizeof(bf16_t);  //  2 MB
    bf16_t* A16 = (bf16_t*)ws;
    bf16_t* W16 = (bf16_t*)(ws + aBytes);
    float* meanv = (float*)(ws + aBytes + wBytes);
    float* varv  = meanv + N_CHUNKS * INP_DIM;

    // 1) convert a and W to bf16
    cvt_bf16_kernel<<<4096, 256, 0, stream>>>(a, A16, B_ROWS * D_A);
    cvt_bf16_kernel<<<1024, 256, 0, stream>>>(W, W16, INP_DIM * D_A);

    // 2) GEMM + bias -> out (as H)
    dim3 gGrid(INP_DIM / 256, B_ROWS / 64);  // (8, 256)
    gemm_bf16_wmma_kernel<<<gGrid, 256, 0, stream>>>(A16, W16, bias, out);

    // 3) ghost BN stats (16 chunks x 2048 cols = 32768 threads)
    gbn_stats_kernel<<<(N_CHUNKS * INP_DIM) / 256, 256, 0, stream>>>(out, meanv,
                                                                    varv);

    // 4) normalize * priors + sparsemax, in-place on out
    sparsemax_kernel<<<B_ROWS, 256, 0, stream>>>(out, priors, meanv, varv,
                                                 gamma, beta);
}
